// GIN_41652592836734
// MI455X (gfx1250) — compile-verified, matrix operand order checked by reading, not run on hardware
//
#include <hip/hip_runtime.h>
#include <hip/hip_bf16.h>

// ---------------------------------------------------------------------------
// Types for WMMA
// ---------------------------------------------------------------------------
typedef __attribute__((ext_vector_type(16))) __bf16         v16bf;
typedef __attribute__((ext_vector_type(8)))  float          v8f;
typedef __attribute__((ext_vector_type(8)))  unsigned short v8us;
typedef __attribute__((ext_vector_type(16))) unsigned short v16us;

union BF16Frag {
    v16us u;
    v16bf b;
};

__device__ __forceinline__ unsigned short f2bf(float f) {
    unsigned int x = __float_as_uint(f);
    // round-to-nearest-even
    unsigned int r = x + 0x7FFFu + ((x >> 16) & 1u);
    return (unsigned short)(r >> 16);
}

// ---------------------------------------------------------------------------
// Weight transpose + fp32 -> bf16 convert.  w: K x N row-major.  wt: N x K.
// ---------------------------------------------------------------------------
__global__ void transpose_cvt_kernel(const float* __restrict__ w,
                                     unsigned short* __restrict__ wt,
                                     int K, int N) {
    int idx = blockIdx.x * blockDim.x + threadIdx.x;
    if (idx >= K * N) return;
    int k = idx / N;
    int n = idx - k * N;
    wt[(size_t)n * K + k] = f2bf(w[idx]);
}

// ---------------------------------------------------------------------------
// CSR build: zero / degree histogram / bucket
// ---------------------------------------------------------------------------
__global__ void zero_i32_kernel(int* __restrict__ p, int n) {
    int i = blockIdx.x * blockDim.x + threadIdx.x;
    if (i < n) p[i] = 0;
}

__global__ void deg_count_kernel(const int* __restrict__ dst,
                                 int* __restrict__ deg, int E) {
    int e = blockIdx.x * blockDim.x + threadIdx.x;
    if (e < E) atomicAdd(&deg[dst[e]], 1);
}

__global__ void bucket_kernel(const int* __restrict__ src,
                              const int* __restrict__ dst,
                              int* __restrict__ cursor,
                              int* __restrict__ csrSrc, int E) {
    int e = blockIdx.x * blockDim.x + threadIdx.x;
    if (e >= E) return;
    int p = atomicAdd(&cursor[dst[e]], 1);
    csrSrc[p] = src[e];
}

// ---------------------------------------------------------------------------
// Prefix sum over degrees (3 phases, LDS block scans)
// ---------------------------------------------------------------------------
__global__ void block_sum_kernel(const int* __restrict__ deg,
                                 int* __restrict__ bsum, int n) {
    __shared__ int s[256];
    int t = threadIdx.x;
    int i = blockIdx.x * 256 + t;
    s[t] = (i < n) ? deg[i] : 0;
    __syncthreads();
    for (int st = 128; st > 0; st >>= 1) {
        if (t < st) s[t] += s[t + st];
        __syncthreads();
    }
    if (t == 0) bsum[blockIdx.x] = s[0];
}

// exclusive scan of nb (<=256) partials, single block of 256
__global__ void scan_partials_kernel(const int* __restrict__ bsum,
                                     int* __restrict__ boff, int nb) {
    __shared__ int s[256];
    int t = threadIdx.x;
    int v0 = (t < nb) ? bsum[t] : 0;
    s[t] = v0;
    __syncthreads();
    for (int st = 1; st < 256; st <<= 1) {
        int u = (t >= st) ? s[t - st] : 0;
        __syncthreads();
        s[t] += u;
        __syncthreads();
    }
    if (t < nb) boff[t] = s[t] - v0;   // exclusive
}

__global__ void scan_block_kernel(const int* __restrict__ deg,
                                  const int* __restrict__ boff,
                                  int* __restrict__ rowStart,
                                  int* __restrict__ cursor, int n) {
    __shared__ int s[256];
    int t = threadIdx.x;
    int i = blockIdx.x * 256 + t;
    int v = (i < n) ? deg[i] : 0;
    s[t] = v;
    __syncthreads();
    for (int st = 1; st < 256; st <<= 1) {
        int u = (t >= st) ? s[t - st] : 0;
        __syncthreads();
        s[t] += u;
        __syncthreads();
    }
    if (i < n) {
        int excl = s[t] - v + boff[blockIdx.x];
        rowStart[i] = excl;
        cursor[i]   = excl;
    }
}

// ---------------------------------------------------------------------------
// Gather aggregation: zb[i] = bf16( feat[i] + sum_{p} feat[csrSrc[p]] )
//   One wave per node; lane holds DPL contiguous floats (D = 32*DPL).
//   Neighbor index is wave-uniform; accumulation stays fp32 in registers.
//   Fuses the GIN self-term and the fp32->bf16 conversion for the next GEMM.
// ---------------------------------------------------------------------------
template <int DPL>   // floats per lane: 4 (D=128) or 8 (D=256)
__global__ void gather_agg_kernel(const float* __restrict__ feat,
                                  const int* __restrict__ csrSrc,
                                  const int* __restrict__ rowStart,
                                  const int* __restrict__ deg,
                                  unsigned short* __restrict__ zb,
                                  int n) {
    const int node = blockIdx.x * (blockDim.x >> 5) + (threadIdx.x >> 5);
    const int lane = threadIdx.x & 31;
    if (node >= n) return;
    constexpr int D = DPL * 32;
    const size_t col = (size_t)lane * DPL;

    float4 acc0 = {0.f, 0.f, 0.f, 0.f};
    float4 acc1 = {0.f, 0.f, 0.f, 0.f};
    {
        const float4* fr = (const float4*)(feat + (size_t)node * D + col);
        acc0 = fr[0];
        if (DPL == 8) acc1 = fr[1];
    }
    const int s0 = rowStart[node];
    const int e0 = s0 + deg[node];
    for (int p = s0; p < e0; ++p) {
        const int s = csrSrc[p];
        const float4* sr = (const float4*)(feat + (size_t)s * D + col);
        float4 v0 = sr[0];
        acc0.x += v0.x; acc0.y += v0.y; acc0.z += v0.z; acc0.w += v0.w;
        if (DPL == 8) {
            float4 v1 = sr[1];
            acc1.x += v1.x; acc1.y += v1.y; acc1.z += v1.z; acc1.w += v1.w;
        }
    }
    ushort4* zr = (ushort4*)(zb + (size_t)node * D + col);
    ushort4 o0 = {f2bf(acc0.x), f2bf(acc0.y), f2bf(acc0.z), f2bf(acc0.w)};
    zr[0] = o0;
    if (DPL == 8) {
        ushort4 o1 = {f2bf(acc1.x), f2bf(acc1.y), f2bf(acc1.z), f2bf(acc1.w)};
        zr[1] = o1;
    }
}

// ---------------------------------------------------------------------------
// WMMA GEMM:  out[M,N] = act( A[M,K] @ W[K,N] + bias )
//   - A is bf16 [M,K] row-major; W pre-transposed bf16 as Wt[N,K]
//   - each wave computes a 16 x (16*NT) strip: one shared A fragment per
//     K-step feeds NT independent WMMAs (A reuse + hazard-free pipelining)
// A-fragment layout (16-bit A 16x32, ISA 7.12.2): lane l holds row (l&15);
// elements 0..7 = K [kb, kb+8), elements 8..15 = K [kb+16, kb+24), kb=8*(l>>4).
// C/D layout: col = lane&15, row = vgpr + 8*(lane>>4).
// ---------------------------------------------------------------------------
template <int RELU, int OUT_BF16, int NT>
__global__ void gemm_wmma_kernel(const unsigned short* __restrict__ A,
                                 const unsigned short* __restrict__ Wt,
                                 const float* __restrict__ bias,
                                 float* __restrict__ outF,
                                 unsigned short* __restrict__ outB,
                                 int N, int K) {
    const int lane = threadIdx.x & 31;
    const int wave = threadIdx.x >> 5;
    const int r    = lane & 15;
    const int hi   = lane >> 4;        // 0 or 1
    const int kb   = hi * 8;
    const int m0   = blockIdx.y * 16;
    const int n0   = (blockIdx.x * (blockDim.x >> 5) + wave) * (16 * NT);

    v8f acc[NT];
#pragma unroll
    for (int t = 0; t < NT; ++t)
        acc[t] = (v8f){0.f, 0.f, 0.f, 0.f, 0.f, 0.f, 0.f, 0.f};

    const unsigned short* arow = A + (size_t)(m0 + r) * K;
    const unsigned short* brow[NT];
#pragma unroll
    for (int t = 0; t < NT; ++t)
        brow[t] = Wt + (size_t)(n0 + t * 16 + r) * K;

    for (int k = 0; k < K; k += 32) {
        BF16Frag af;
        {
            v8us a0 = *(const v8us*)(arow + k + kb);
            v8us a1 = *(const v8us*)(arow + k + kb + 16);
            af.u = __builtin_shufflevector(a0, a1,
                    0, 1, 2, 3, 4, 5, 6, 7, 8, 9, 10, 11, 12, 13, 14, 15);
        }
#pragma unroll
        for (int t = 0; t < NT; ++t) {
            BF16Frag bf;
            v8us b0 = *(const v8us*)(brow[t] + k + kb);
            v8us b1 = *(const v8us*)(brow[t] + k + kb + 16);
            bf.u = __builtin_shufflevector(b0, b1,
                    0, 1, 2, 3, 4, 5, 6, 7, 8, 9, 10, 11, 12, 13, 14, 15);
            acc[t] = __builtin_amdgcn_wmma_f32_16x16x32_bf16(
                false, af.b, false, bf.b, (short)0, acc[t], false, false);
        }
    }

#pragma unroll
    for (int t = 0; t < NT; ++t) {
        const float bs = bias[n0 + t * 16 + r];
#pragma unroll
        for (int v = 0; v < 8; ++v) {
            int row   = m0 + v + 8 * hi;
            float val = acc[t][v] + bs;
            if (RELU) val = fmaxf(val, 0.0f);
            size_t o = (size_t)row * N + n0 + t * 16 + r;
            if (OUT_BF16) outB[o] = f2bf(val);
            else          outF[o] = val;
        }
    }
}

// ---------------------------------------------------------------------------
// In-place log_softmax over rows of 16
// ---------------------------------------------------------------------------
__global__ void log_softmax16_kernel(float* __restrict__ out, int M) {
    int i = blockIdx.x * blockDim.x + threadIdx.x;
    if (i >= M) return;
    float* row = out + (size_t)i * 16;
    float v[16];
    float mx = -3.402823466e+38f;
#pragma unroll
    for (int j = 0; j < 16; ++j) { v[j] = row[j]; mx = fmaxf(mx, v[j]); }
    float s = 0.f;
#pragma unroll
    for (int j = 0; j < 16; ++j) s += __expf(v[j] - mx);
    float lse = mx + __logf(s);
#pragma unroll
    for (int j = 0; j < 16; ++j) row[j] = v[j] - lse;
}

// ---------------------------------------------------------------------------
// Launch
// ---------------------------------------------------------------------------
extern "C" void kernel_launch(void* const* d_in, const int* in_sizes, int n_in,
                              void* d_out, int out_size, void* d_ws, size_t ws_size,
                              hipStream_t stream) {
    constexpr int NN = 50000, NE = 600000;
    constexpr int IN = 128, HID = 256, OUT = 16;
    constexpr int NB = (NN + 255) / 256;   // 196 partial blocks (<= 256)

    const float* x   = (const float*)d_in[0];
    const int*   ei  = (const int*)d_in[1];       // [2, NE]: src then dst
    const float* w1a = (const float*)d_in[2];
    const float* b1a = (const float*)d_in[3];
    const float* w2a = (const float*)d_in[4];
    const float* b2a = (const float*)d_in[5];
    const float* w1b = (const float*)d_in[6];
    const float* b1b = (const float*)d_in[7];
    const float* w2b = (const float*)d_in[8];
    const float* b2b = (const float*)d_in[9];
    float* out = (float*)d_out;

    const int* src = ei;
    const int* dst = ei + NE;

    // workspace carve-out (256B aligned)
    char* ws = (char*)d_ws;
    size_t off = 0;
    auto carve = [&](size_t bytes) -> char* {
        char* p = ws + off;
        off += (bytes + 255) & ~(size_t)255;
        return p;
    };
    unsigned short* wt1a = (unsigned short*)carve((size_t)HID * IN  * 2);
    unsigned short* wt2a = (unsigned short*)carve((size_t)HID * HID * 2);
    unsigned short* wt1b = (unsigned short*)carve((size_t)HID * HID * 2);
    unsigned short* wt2b = (unsigned short*)carve((size_t)OUT * HID * 2);
    unsigned short* zb   = (unsigned short*)carve((size_t)NN * HID * 2); // z bf16
    unsigned short* tbuf = (unsigned short*)carve((size_t)NN * HID * 2); // t bf16
    float*          hbuf = (float*)carve((size_t)NN * HID * 4);          // h1 fp32
    int* deg      = (int*)carve((size_t)NN * 4);
    int* rowStart = (int*)carve((size_t)NN * 4);
    int* cursor   = (int*)carve((size_t)NN * 4);
    int* bsum     = (int*)carve((size_t)256 * 4);
    int* boff     = (int*)carve((size_t)256 * 4);
    int* csrSrc   = (int*)carve((size_t)NE * 4);

    // 1) weights -> bf16, transposed to [N,K]
    transpose_cvt_kernel<<<(IN * HID + 255) / 256, 256, 0, stream>>>(w1a, wt1a, IN, HID);
    transpose_cvt_kernel<<<(HID * HID + 255) / 256, 256, 0, stream>>>(w2a, wt2a, HID, HID);
    transpose_cvt_kernel<<<(HID * HID + 255) / 256, 256, 0, stream>>>(w1b, wt1b, HID, HID);
    transpose_cvt_kernel<<<(HID * OUT + 255) / 256, 256, 0, stream>>>(w2b, wt2b, HID, OUT);

    // 2) CSR-by-destination build
    zero_i32_kernel<<<(NN + 255) / 256, 256, 0, stream>>>(deg, NN);
    deg_count_kernel<<<(NE + 255) / 256, 256, 0, stream>>>(dst, deg, NE);
    block_sum_kernel<<<NB, 256, 0, stream>>>(deg, bsum, NN);
    scan_partials_kernel<<<1, 256, 0, stream>>>(bsum, boff, NB);
    scan_block_kernel<<<NB, 256, 0, stream>>>(deg, boff, rowStart, cursor, NN);
    bucket_kernel<<<(NE + 255) / 256, 256, 0, stream>>>(src, dst, cursor, csrSrc, NE);

    // 3) zb = bf16( x + gather-sum(x) )     [wave per node, 8 nodes/block]
    gather_agg_kernel<4><<<(NN + 7) / 8, 256, 0, stream>>>(
        x, csrSrc, rowStart, deg, zb, NN);

    // 4) t1 = relu(z1 @ w1a + b1a)          [bf16 -> bf16], 4 waves x (16x64)
    gemm_wmma_kernel<1, 1, 4><<<dim3(1, NN / 16), 128, 0, stream>>>(
        zb, wt1a, b1a, nullptr, tbuf, HID, IN);

    // 5) h1 = relu(t1 @ w2a + b2a)          [bf16 -> fp32; fuses outer relu]
    gemm_wmma_kernel<1, 0, 4><<<dim3(1, NN / 16), 128, 0, stream>>>(
        tbuf, wt2a, b2a, hbuf, nullptr, HID, HID);

    // 6) zb = bf16( h1 + gather-sum(h1) )
    gather_agg_kernel<8><<<(NN + 7) / 8, 256, 0, stream>>>(
        hbuf, csrSrc, rowStart, deg, zb, NN);

    // 7) t2 = relu(z2 @ w1b + b1b)          [bf16 -> bf16]
    gemm_wmma_kernel<1, 1, 4><<<dim3(1, NN / 16), 128, 0, stream>>>(
        zb, wt1b, b1b, nullptr, tbuf, HID, HID);

    // 8) logits = t2 @ w2b + b2b            [bf16 -> fp32 into d_out]
    gemm_wmma_kernel<0, 0, 1><<<dim3(1, NN / 16), 32, 0, stream>>>(
        tbuf, wt2b, b2b, out, nullptr, OUT, HID);

    // 9) in-place log_softmax
    log_softmax16_kernel<<<(NN + 255) / 256, 256, 0, stream>>>(out, NN);
}
